// GraphAttentionLayer_50586124812299
// MI455X (gfx1250) — compile-verified, matrix operand order checked by reading, not run on hardware
//
#include <hip/hip_runtime.h>

#define BB 4
#define NN 4096
#define DIN 256
#define DOUT 256
#define HEADS 4
#define DK 64

typedef __attribute__((ext_vector_type(16))) __bf16 v16bf;
typedef __attribute__((ext_vector_type(8)))  __bf16 v8bf;
typedef __attribute__((ext_vector_type(8)))  float  v8f;
typedef __attribute__((ext_vector_type(4)))  int    v4i;

// Load a 16-element bf16 WMMA fragment slice for this lane.
// `p` points at the start of a contiguous 32-element chunk along the K dim.
// A-layout (lane holds row lane%16) and B-layout (lane holds col lane%16)
// use the identical per-lane K pattern: {8h..8h+7} and {16+8h..16+8h+7}.
static __device__ inline v16bf load_frag(const __bf16* p, int h) {
  const v8bf* q = (const v8bf*)p;
  v8bf lo = q[h];
  v8bf hi = q[2 + h];
  v16bf f;
#pragma unroll
  for (int i = 0; i < 8; ++i) { f[i] = lo[i]; f[i + 8] = hi[i]; }
  return f;
}

static __device__ inline v8f wmma_bf16(v16bf a, v16bf b, v8f c) {
  return __builtin_amdgcn_wmma_f32_16x16x32_bf16(false, a, false, b, (short)0, c,
                                                 false, false);
}

// ---------- fp32 -> bf16 elementwise convert ----------
__global__ void cvt_kernel(const float* __restrict__ in, __bf16* __restrict__ out, int n) {
  int i = blockIdx.x * blockDim.x + threadIdx.x;
  if (i < n) out[i] = (__bf16)in[i];
}

// ---------- fp32 W[in][out] -> bf16 Wt[out][in] ----------
__global__ void tcvt_kernel(const float* __restrict__ W, __bf16* __restrict__ Wt) {
  int i = blockIdx.x * 16 + threadIdx.x;  // in dim
  int o = blockIdx.y * 16 + threadIdx.y;  // out dim
  Wt[(size_t)o * DIN + i] = (__bf16)W[(size_t)i * DOUT + o];
}

// ---------- QKV projection: one wave -> 16 tokens x 16 out-dims, 3 mats ----------
__global__ void qkv_kernel(const __bf16* __restrict__ xb,
                           const __bf16* __restrict__ wqt, const __bf16* __restrict__ wkt,
                           const __bf16* __restrict__ wvt,
                           const float* __restrict__ bq, const float* __restrict__ bk,
                           const float* __restrict__ bv,
                           __bf16* __restrict__ qb, __bf16* __restrict__ kb,
                           __bf16* __restrict__ vt) {
  int lane = threadIdx.x & 31;
  int h    = lane >> 4;
  int col  = lane & 15;
  int ttile = blockIdx.x;
  int otile = blockIdx.y;
  int od = otile * 16 + col;

  const __bf16* xrow = xb + (size_t)(ttile * 16 + col) * DIN;  // A row (M = col)
  const __bf16* wq = wqt + (size_t)od * DIN;                   // B col (n = col)
  const __bf16* wk = wkt + (size_t)od * DIN;
  const __bf16* wv = wvt + (size_t)od * DIN;

  v8f cq = {}, ck = {}, cv = {};
#pragma unroll
  for (int c = 0; c < DIN / 32; ++c) {
    v16bf a  = load_frag(xrow + c * 32, h);
    v16bf b0 = load_frag(wq + c * 32, h);
    v16bf b1 = load_frag(wk + c * 32, h);
    v16bf b2 = load_frag(wv + c * 32, h);
    cq = wmma_bf16(a, b0, cq);
    ck = wmma_bf16(a, b1, ck);
    cv = wmma_bf16(a, b2, cv);
  }
  float biasq = bq[od], biask = bk[od], biasv = bv[od];
  int head = od >> 6, dloc = od & 63;
#pragma unroll
  for (int r = 0; r < 8; ++r) {
    int tok = ttile * 16 + r + 8 * h;   // global token (row m)
    qb[(size_t)tok * DOUT + od] = (__bf16)(cq[r] + biasq);
    kb[(size_t)tok * DOUT + od] = (__bf16)(ck[r] + biask);
    int bi = tok / NN, nloc = tok % NN;
    // V stored transposed per head: Vt[b][h][d][n]
    vt[((size_t)(bi * HEADS + head) * DK + dloc) * NN + nloc] = (__bf16)(cv[r] + biasv);
  }
}

// ---------- fused masked flash attention, Sᵀ = K·Qᵀ, Oᵀ += Vᵀ·Pᵀ ----------
// block = 128 threads = 4 waves; wave w handles head w of one 16-query tile.
// adj (the dominant 256MB stream) is staged once per block into LDS with
// async global->LDS copies (double buffered, ASYNCcnt), shared by all 4 heads.
__global__ void attn_kernel(const __bf16* __restrict__ qb, const __bf16* __restrict__ kbuf,
                            const __bf16* __restrict__ vt, const int* __restrict__ adj,
                            __bf16* __restrict__ ao) {
  // rows padded to 36 ints (144B): keeps 16B alignment for ds_load_b128,
  // spreads the 16 query rows across LDS banks.
  __shared__ __align__(16) int adjs[2][16][36];

  int tid  = threadIdx.x;
  int lane = tid & 31;
  int head = tid >> 5;
  int h    = lane >> 4;
  int col  = lane & 15;
  int bt = blockIdx.x;
  int b     = bt / (NN / 16);
  int qtile = bt % (NN / 16);
  int qloc = qtile * 16 + col;          // this lane's query (column of Sᵀ)
  size_t tokq = (size_t)b * NN + qloc;

  // cooperative async adj staging: 128 threads x 16B = 2KB per 32-key chunk
  int lrow  = tid >> 3;                 // 0..15 query row
  int lelem = (tid & 7) * 4;            // 0,4,...,28 key offset
  const int* agbase = adj + ((size_t)b * NN + qtile * 16 + lrow) * (size_t)NN + lelem;

  // prefetch chunk 0 into buffer 0
  {
    unsigned ldsa = (unsigned)(unsigned long long)&adjs[0][lrow][lelem];
    unsigned long long g = (unsigned long long)agbase;
    asm volatile("global_load_async_to_lds_b128 %0, %1, off th:TH_LOAD_NT"
                 :: "v"(ldsa), "v"(g) : "memory");
  }

  // Q B-fragments (d_k = 64 -> 2 chunks of 32)
  const __bf16* qrow = qb + tokq * DOUT + head * DK;
  v16bf qf0 = load_frag(qrow, h);
  v16bf qf1 = load_frag(qrow + 32, h);

  const float scale = 0.125f;           // 1/sqrt(64 + 1e-9)
  float mrun = -1e30f, lrun = 0.f;
  v8f acc[4] = {};                      // Oᵀ tiles: 4 x (16 d x 16 q)

  const __bf16* vbase = vt + (size_t)(b * HEADS + head) * DK * NN;
  const __bf16* kbase = kbuf + (size_t)b * NN * DOUT + head * DK;

  for (int k0 = 0; k0 < NN; k0 += 32) {
    int cb = (k0 >> 5) & 1;
    // issue prefetch of next chunk into the other buffer, then wait for ours
    if (k0 + 32 < NN) {
      unsigned ldsa = (unsigned)(unsigned long long)&adjs[cb ^ 1][lrow][lelem];
      unsigned long long g = (unsigned long long)(agbase + k0 + 32);
      asm volatile("global_load_async_to_lds_b128 %0, %1, off th:TH_LOAD_NT"
                   :: "v"(ldsa), "v"(g) : "memory");
      asm volatile("s_wait_asynccnt 0x1" ::: "memory");   // our chunk (older) done
    } else {
      asm volatile("s_wait_asynccnt 0x0" ::: "memory");
    }
    __syncthreads();                    // all threads' portions visible

    // --- two 16-key subtiles of Sᵀ ---
    v8f s0 = {}, s1 = {};
    {
      const __bf16* kr = kbase + (size_t)(k0 + col) * DOUT;       // A row: key k0+col
      s0 = wmma_bf16(load_frag(kr, h), qf0, s0);
      s0 = wmma_bf16(load_frag(kr + 32, h), qf1, s0);
    }
    {
      const __bf16* kr = kbase + (size_t)(k0 + 16 + col) * DOUT;
      s1 = wmma_bf16(load_frag(kr, h), qf0, s1);
      s1 = wmma_bf16(load_frag(kr + 32, h), qf1, s1);
    }

    // --- mask via adj | eye, read from LDS (ds_load_b128 x4) ---
    v4i A00 = *(const v4i*)&adjs[cb][col][8 * h];
    v4i A01 = *(const v4i*)&adjs[cb][col][8 * h + 4];
    v4i A10 = *(const v4i*)&adjs[cb][col][16 + 8 * h];
    v4i A11 = *(const v4i*)&adjs[cb][col][16 + 8 * h + 4];

    float sc[16];
    float lmax = -1e30f;
#pragma unroll
    for (int r = 0; r < 8; ++r) {
      int key0 = k0 + r + 8 * h;
      int a0 = (r < 4) ? A00[r] : A01[r - 4];
      float u0 = (a0 != 0 || key0 == qloc) ? s0[r] * scale : -1e30f;
      sc[r] = u0; lmax = fmaxf(lmax, u0);
      int key1 = k0 + 16 + r + 8 * h;
      int a1 = (r < 4) ? A10[r] : A11[r - 4];
      float u1 = (a1 != 0 || key1 == qloc) ? s1[r] * scale : -1e30f;
      sc[8 + r] = u1; lmax = fmaxf(lmax, u1);
    }
    __syncthreads();                    // done reading this buffer

    // merge the two lane-halves that share a query column
    lmax = fmaxf(lmax, __shfl_xor(lmax, 16, 32));
    float mnew  = fmaxf(mrun, lmax);
    float alpha = __expf(mrun - mnew);
    float lsum = 0.f;
    v16bf pf;                           // P packed directly as a B-fragment
#pragma unroll
    for (int i = 0; i < 16; ++i) {
      float p = __expf(sc[i] - mnew);
      lsum += p;
      pf[i] = (__bf16)p;
    }
    lsum += __shfl_xor(lsum, 16, 32);
    lrun = lrun * alpha + lsum;
    mrun = mnew;
#pragma unroll
    for (int t = 0; t < 4; ++t)
#pragma unroll
      for (int r = 0; r < 8; ++r) acc[t][r] *= alpha;

    // --- Oᵀ += Vᵀ · Pᵀ : 4 d-tiles, k = 32 keys ---
#pragma unroll
    for (int t = 0; t < 4; ++t) {
      const __bf16* vr = vbase + (size_t)(t * 16 + col) * NN + k0;  // A row: dim d
      acc[t] = wmma_bf16(load_frag(vr, h), pf, acc[t]);
    }
  }

  float inv = (lrun > 0.f) ? 1.f / lrun : 0.f;
#pragma unroll
  for (int t = 0; t < 4; ++t)
#pragma unroll
    for (int r = 0; r < 8; ++r) {
      int d = t * 16 + r + 8 * h;
      ao[tokq * DOUT + head * DK + d] = (__bf16)(acc[t][r] * inv);
    }
}

// ---------- output projection + bias + relu ----------
__global__ void oproj_kernel(const __bf16* __restrict__ ao, const __bf16* __restrict__ wot,
                             const float* __restrict__ bo, float* __restrict__ out) {
  int lane = threadIdx.x & 31;
  int h = lane >> 4, col = lane & 15;
  int ttile = blockIdx.x, otile = blockIdx.y;
  int od = otile * 16 + col;
  const __bf16* arow = ao + (size_t)(ttile * 16 + col) * DOUT;
  const __bf16* wrow = wot + (size_t)od * DOUT;
  v8f c = {};
#pragma unroll
  for (int ch = 0; ch < DOUT / 32; ++ch)
    c = wmma_bf16(load_frag(arow + ch * 32, h), load_frag(wrow + ch * 32, h), c);
  float bias = bo[od];
#pragma unroll
  for (int r = 0; r < 8; ++r) {
    int tok = ttile * 16 + r + 8 * h;
    float v = c[r] + bias;
    out[(size_t)tok * DOUT + od] = v > 0.f ? v : 0.f;
  }
}

extern "C" void kernel_launch(void* const* d_in, const int* in_sizes, int n_in,
                              void* d_out, int out_size, void* d_ws, size_t ws_size,
                              hipStream_t stream) {
  const float* x  = (const float*)d_in[0];
  const int*   adj= (const int*)d_in[1];
  const float* Wq = (const float*)d_in[2];
  const float* bq = (const float*)d_in[3];
  const float* Wk = (const float*)d_in[4];
  const float* bk = (const float*)d_in[5];
  const float* Wv = (const float*)d_in[6];
  const float* bv = (const float*)d_in[7];
  const float* Wo = (const float*)d_in[8];
  const float* bo = (const float*)d_in[9];
  float* out = (float*)d_out;

  char* ws = (char*)d_ws;
  const size_t SZ  = (size_t)BB * NN * DOUT * sizeof(__bf16);   // 8 MB
  const size_t WSZ = (size_t)DIN * DOUT * sizeof(__bf16);       // 128 KB
  __bf16* xb  = (__bf16*)(ws);
  __bf16* qb  = (__bf16*)(ws + SZ);
  __bf16* kbf = (__bf16*)(ws + 2 * SZ);
  __bf16* vt  = (__bf16*)(ws + 3 * SZ);
  __bf16* ao  = (__bf16*)(ws + 4 * SZ);
  __bf16* wqt = (__bf16*)(ws + 5 * SZ);
  __bf16* wkt = (__bf16*)(ws + 5 * SZ + WSZ);
  __bf16* wvt = (__bf16*)(ws + 5 * SZ + 2 * WSZ);
  __bf16* wot = (__bf16*)(ws + 5 * SZ + 3 * WSZ);

  int nx = BB * NN * DIN;
  cvt_kernel<<<nx / 256, 256, 0, stream>>>(x, xb, nx);
  dim3 tb(16, 16), tg(DIN / 16, DOUT / 16);
  tcvt_kernel<<<tg, tb, 0, stream>>>(Wq, wqt);
  tcvt_kernel<<<tg, tb, 0, stream>>>(Wk, wkt);
  tcvt_kernel<<<tg, tb, 0, stream>>>(Wv, wvt);
  tcvt_kernel<<<tg, tb, 0, stream>>>(Wo, wot);

  dim3 gp(BB * NN / 16, DOUT / 16);
  qkv_kernel<<<gp, 32, 0, stream>>>(xb, wqt, wkt, wvt, bq, bk, bv, qb, kbf, vt);
  attn_kernel<<<BB * NN / 16, 128, 0, stream>>>(qb, kbf, vt, adj, ao);
  oproj_kernel<<<gp, 32, 0, stream>>>(ao, wot, bo, out);
}
